// Setr_ConvFormer_73254962200658
// MI455X (gfx1250) — matrix operand order repaired and checked by reading, not compiled, not run on hardware
//
#include <hip/hip_runtime.h>

#define DEV __device__ __forceinline__

typedef __bf16 bf16;
typedef unsigned u32;
typedef bf16  v8bf  __attribute__((ext_vector_type(8)));
typedef bf16  v16bf __attribute__((ext_vector_type(16)));
typedef float v4f   __attribute__((ext_vector_type(4)));
typedef float v8f   __attribute__((ext_vector_type(8)));

DEV bf16 f2bf(float f) {
  u32 u = __builtin_bit_cast(u32, f);
  u += 0x7FFFu + ((u >> 16) & 1u);               // RNE
  unsigned short h = (unsigned short)(u >> 16);
  union { unsigned short s; bf16 b; } c; c.s = h;
  return c.b;
}

// pack two floats -> two bf16 in one dword (lo = a, hi = b), RNE
DEV u32 packbf2(float a, float b) {
  u32 ua = __builtin_bit_cast(u32, a);
  u32 ub = __builtin_bit_cast(u32, b);
  ua += 0x7FFFu + ((ua >> 16) & 1u);
  ub += 0x7FFFu + ((ub >> 16) & 1u);
  return (ua >> 16) | (ub & 0xFFFF0000u);
}

// A fragment 16x32 bf16 from a row-major [16][32] bf16 tile ([16][16] u32).
// Lane(hi,lo): row=lo, K = hi*16..+15 -> 2x ds_load_b128.
DEV v16bf afrag16x32(const u32* tile, int lane) {
  int hi = (lane >> 4) & 1, lo = lane & 15;
  const v8bf* p = (const v8bf*)(tile + lo * 16);
  v8bf a0 = p[2 * hi];
  v8bf a1 = p[2 * hi + 1];
  return __builtin_shufflevector(a0, a1, 0, 1, 2, 3, 4, 5, 6, 7,
                                         8, 9, 10, 11, 12, 13, 14, 15);
}

// B fragment 32x16 bf16 from a TRANSPOSED tile BsT[16 n][32 k] ([16][16] u32).
// Lane(hi,lo): col=lo, K = hi*16..+15 -> 16 contiguous bf16 -> 2x ds_load_b128.
DEV v16bf bfrag32x16(const u32* tileT, int lane) {
  int hi = (lane >> 4) & 1, lo = lane & 15;
  const v8bf* p = (const v8bf*)(tileT + lo * 16 + hi * 8);
  return __builtin_shufflevector(p[0], p[1], 0, 1, 2, 3, 4, 5, 6, 7,
                                             8, 9, 10, 11, 12, 13, 14, 15);
}

// shared epilogue: BN+ReLU (+residual), 8 consecutive m per lane -> 2x b128
DEV void store_tile(const v8f& acc, float* out, const float* res,
                    const float* bg, const float* bb, const float* bm, const float* bv,
                    int n, size_t base) {
  float scale = 1.f, shift = 0.f;
  bool bn = (bg != nullptr);
  if (bn) {
    float inv = rsqrtf(bv[n] + 1e-5f);
    scale = bg[n] * inv;
    shift = bb[n] - bm[n] * scale;
  }
  v4f o0, o1;
#pragma unroll
  for (int r = 0; r < 4; ++r) {
    float u = acc[r], v = acc[4 + r];
    if (bn) {
      u = u * scale + shift; u = u > 0.f ? u : 0.f;
      v = v * scale + shift; v = v > 0.f ? v : 0.f;
    }
    o0[r] = u; o1[r] = v;
  }
  if (res) {
    const v4f* rp = (const v4f*)(res + base);
    o0 += rp[0]; o1 += rp[1];
  }
  v4f* op = (v4f*)(out + base);
  op[0] = o0; op[1] = o1;
}

// ---------------- conv3x3 (pad=1) implicit-GEMM, Cin % 32 == 0 ---------------
// k = rs*Cin + c ordering: the 3x3 tap (dy,dx) is uniform per 32-wide k-chunk.
// Block: 128 thr = 4 waves; tile 64 m x (NT*16) n; NT WMMAs per wave per chunk.
template <int NT>
__global__ __launch_bounds__(128)
void conv3x3_wmma(const float* __restrict__ in, const float* __restrict__ w,
                  const float* bg, const float* bb, const float* bm, const float* bv,
                  const float* __restrict__ res, float* __restrict__ out,
                  int Cin, int Cout, int H, int W, int Wlog2) {
  const int HW = H * W;
  const int K  = Cin * 9;
  const int b  = blockIdx.z;
  const int n0 = blockIdx.y * (16 * NT);
  const int m0 = blockIdx.x * 64;
  const int tid = threadIdx.x;
  const int wv = tid >> 5;
  const int lane = tid & 31;

  __shared__ __align__(16) u32 As32[64 * 16];        // [64 m][32 k] bf16
  __shared__ __align__(16) u32 BsT32[NT * 16 * 16];  // NT x [16 n][32 k] bf16

  v8f acc[NT];
#pragma unroll
  for (int t = 0; t < NT; ++t) acc[t] = (v8f){};

  const float* inB = in + (size_t)b * Cin * HW;
  const int mi_f = tid & 63;
  const int kph  = tid >> 6;
  const int m_f  = m0 + mi_f;
  const int y0_f = m_f >> Wlog2;
  const int x0_f = m_f & (W - 1);

  for (int kc = 0; kc < K; kc += 32) {
    if (((tid & 63) == 0) && (kc + 32 < K))
      __builtin_prefetch(&w[(size_t)n0 * K + kc + 32], 0, 3);  // global_prefetch_b8

    const int rs = kc / Cin;            // uniform tap for this chunk
    const int cb = kc % Cin;
    const int dy = rs / 3 - 1, dx = rs % 3 - 1;
    const int y = y0_f + dy, x = x0_f + dx;
    const bool ok = (y >= 0) & (y < H) & (x >= 0) & (x < W);
    const float* src = inB + ((size_t)cb * H + y) * W + x;     // +c -> +HW
#pragma unroll
    for (int it = 0; it < 8; ++it) {
      int kp = kph + 2 * it;            // pair index 0..15
      float f0 = 0.f, f1 = 0.f;
      if (ok) {
        f0 = src[(size_t)(2 * kp) * HW];
        f1 = src[(size_t)(2 * kp + 1) * HW];
      }
      As32[mi_f * 16 + kp] = packbf2(f0, f1);
    }
#pragma unroll
    for (int it = 0; it < 2 * NT; ++it) {
      int p = tid + 128 * it;           // NT*256 pairs
      int t = p >> 8;
      int n = (p >> 4) & 15, kp = p & 15;
      size_t wb = (size_t)(n0 + t * 16 + n) * K + (size_t)(cb + 2 * kp) * 9 + rs;
      BsT32[p] = packbf2(w[wb], w[wb + 9]);
    }
    __syncthreads();
    v16bf a = afrag16x32(As32 + wv * 256, lane);
#pragma unroll
    for (int t = 0; t < NT; ++t) {
      v16bf bf = bfrag32x16(BsT32 + t * 256, lane);
      acc[t] = __builtin_amdgcn_wmma_f32_16x16x32_bf16(false, a, false, bf, (short)0,
                                                       acc[t], false, false);
    }
    __syncthreads();
  }

  int hi = (lane >> 4) & 1, lo = lane & 15;
#pragma unroll
  for (int t = 0; t < NT; ++t) {
    int n = n0 + t * 16 + lo;
    size_t base = ((size_t)b * Cout + n) * HW + m0 + wv * 16 + hi * 8;
    store_tile(acc[t], out, res, bg, bb, bm, bv, n, base);
  }
}

// ---------------- conv3x3 generic (small Cin, e.g. 3) ------------------------
__global__ __launch_bounds__(128)
void conv3x3_small(const float* __restrict__ in, const float* __restrict__ w,
                   const float* bg, const float* bb, const float* bm, const float* bv,
                   float* __restrict__ out,
                   int Cin, int Cout, int H, int W, int Wlog2) {
  const int HW = H * W;
  const int K  = Cin * 9;
  const int b  = blockIdx.z;
  const int n0 = blockIdx.y * 16;
  const int m0 = blockIdx.x * 64;
  const int tid = threadIdx.x;
  const int wv = tid >> 5;
  const int lane = tid & 31;

  __shared__ __align__(16) u32 As32[64 * 16];
  __shared__ __align__(16) u32 BsT32[16 * 16];

  v8f acc = {};
  const float* inB = in + (size_t)b * Cin * HW;
  bf16* AsH  = (bf16*)As32;
  bf16* BsTH = (bf16*)BsT32;

  for (int kc = 0; kc < K; kc += 32) {
    for (int idx = tid; idx < 64 * 32; idx += 128) {
      int mi = idx >> 5, kk = idx & 31;
      int gk = kc + kk;
      float val = 0.f;
      if (gk < K) {
        int c = gk / 9, rs = gk % 9;
        int dy = rs / 3 - 1, dx = rs % 3 - 1;
        int m = m0 + mi;
        int y = (m >> Wlog2) + dy, x = (m & (W - 1)) + dx;
        if (y >= 0 && y < H && x >= 0 && x < W)
          val = inB[((size_t)c * H + y) * W + x];
      }
      AsH[mi * 32 + kk] = f2bf(val);
    }
    for (int idx = tid; idx < 16 * 32; idx += 128) {
      int n = idx >> 5, kk = idx & 31;
      int gk = kc + kk;
      float val = (gk < K) ? w[(size_t)(n0 + n) * K + gk] : 0.f;
      BsTH[n * 32 + kk] = f2bf(val);
    }
    __syncthreads();
    v16bf a = afrag16x32(As32 + wv * 256, lane);
    v16bf bf = bfrag32x16(BsT32, lane);
    acc = __builtin_amdgcn_wmma_f32_16x16x32_bf16(false, a, false, bf, (short)0, acc, false, false);
    __syncthreads();
  }

  int hi = (lane >> 4) & 1, lo = lane & 15;
  int n = n0 + lo;
  size_t base = ((size_t)b * Cout + n) * HW + m0 + wv * 16 + hi * 8;
  store_tile(acc, out, nullptr, bg, bb, bm, bv, n, base);
}

// ---------------- conv1x1 GEMM, NT n-tiles per wave --------------------------
template <int NT>
__global__ __launch_bounds__(128)
void conv1x1_wmma(const float* __restrict__ in, const float* __restrict__ w,
                  const float* bg, const float* bb, const float* bm, const float* bv,
                  const float* __restrict__ res, float* __restrict__ out,
                  int Cin, int Cout, int HW) {
  const int b  = blockIdx.z;
  const int n0 = blockIdx.y * (16 * NT);
  const int m0 = blockIdx.x * 64;
  const int tid = threadIdx.x;
  const int wv = tid >> 5;
  const int lane = tid & 31;

  __shared__ __align__(16) u32 As32[64 * 16];
  __shared__ __align__(16) u32 BsT32[NT * 16 * 16];

  v8f acc[NT];
#pragma unroll
  for (int t = 0; t < NT; ++t) acc[t] = (v8f){};

  const float* inB = in + (size_t)b * Cin * HW;
  const int mi_f = tid & 63;
  const int kph  = tid >> 6;
  const float* srcm = inB + m0 + mi_f;

  for (int kc = 0; kc < Cin; kc += 32) {
#pragma unroll
    for (int it = 0; it < 8; ++it) {
      int kp = kph + 2 * it;
      float f0 = srcm[(size_t)(kc + 2 * kp) * HW];
      float f1 = srcm[(size_t)(kc + 2 * kp + 1) * HW];
      As32[mi_f * 16 + kp] = packbf2(f0, f1);
    }
#pragma unroll
    for (int it = 0; it < 2 * NT; ++it) {
      int p = tid + 128 * it;
      int t = p >> 8;
      int n = (p >> 4) & 15, kp = p & 15;
      const float* wp = w + (size_t)(n0 + t * 16 + n) * Cin + kc + 2 * kp;  // contiguous
      BsT32[p] = packbf2(wp[0], wp[1]);
    }
    __syncthreads();
    v16bf a = afrag16x32(As32 + wv * 256, lane);
#pragma unroll
    for (int t = 0; t < NT; ++t) {
      v16bf bf = bfrag32x16(BsT32 + t * 256, lane);
      acc[t] = __builtin_amdgcn_wmma_f32_16x16x32_bf16(false, a, false, bf, (short)0,
                                                       acc[t], false, false);
    }
    __syncthreads();
  }

  int hi = (lane >> 4) & 1, lo = lane & 15;
#pragma unroll
  for (int t = 0; t < NT; ++t) {
    int n = n0 + t * 16 + lo;
    size_t base = ((size_t)b * Cout + n) * HW + m0 + wv * 16 + hi * 8;
    store_tile(acc[t], out, res, bg, bb, bm, bv, n, base);
  }
}

// ---------------- maxpool 2x2 stride 2 ---------------------------------------
__global__ void maxpool2(const float* __restrict__ in, float* __restrict__ out,
                         int C, int H, int W, int total) {
  int i = blockIdx.x * blockDim.x + threadIdx.x;
  if (i >= total) return;
  int Wo = W >> 1, Ho = H >> 1;
  int x = i % Wo; int t = i / Wo;
  int y = t % Ho; t /= Ho;
  int c = t % C; int b = t / C;
  const float* p = in + (((size_t)b * C + c) * H + 2 * y) * W + 2 * x;
  out[i] = fmaxf(fmaxf(p[0], p[1]), fmaxf(p[W], p[W + 1]));
}

// ---------------- q/k norms:  sqrt(sum_d t^2 + 1e-4) -------------------------
__global__ void qk_norms(const float* __restrict__ qkv, float* __restrict__ nq,
                         float* __restrict__ nk, int B, int N) {
  int i = blockIdx.x * blockDim.x + threadIdx.x;
  if (i >= 2 * B * N) return;
  int qk = i / (B * N);
  int rem = i % (B * N);
  int b = rem / N, n = rem % N;
  int cbase = qk ? 32 : 0;
  const float* p = qkv + ((size_t)b * 96 + cbase) * N + n;
  float s = 0.f;
#pragma unroll
  for (int d = 0; d < 32; ++d) { float v = p[(size_t)d * N]; s += v * v; }
  (qk ? nk : nq)[(size_t)b * N + n] = sqrtf(s + 1e-4f);
}

// ---------------- cosine attention, flash-style, 1 wave / 16 rows ------------
__global__ __launch_bounds__(32)
void attention_wmma(const float* __restrict__ qkv, const float* __restrict__ nq,
                    const float* __restrict__ nk, float* __restrict__ outp, int N) {
  const int b = blockIdx.y;
  const int n0 = blockIdx.x * 16;
  const int lane = threadIdx.x & 31;
  const int hi = (lane >> 4) & 1, lo = lane & 15;
  __shared__ __align__(16) bf16 Sbuf[16 * 32];

  const float* qb = qkv + (size_t)b * 96 * N;

  union { v16bf v; u32 u[8]; } QA;
#pragma unroll
  for (int j = 0; j < 8; ++j) {
    int kb = hi * 16 + 2 * j;
    const float* c0 = qb + (size_t)kb * N + n0 + lo;
    QA.u[j] = packbf2(c0[0], c0[N]);
  }
  float qnr[8];
#pragma unroll
  for (int r = 0; r < 8; ++r) qnr[r] = nq[(size_t)b * N + n0 + hi * 8 + r];

  v8f O0 = {}, O1 = {};
  for (int mb = 0; mb < N; mb += 32) {
    union { v16bf v; u32 u[8]; } K0, K1;
#pragma unroll
    for (int j = 0; j < 8; ++j) {
      int d = hi * 16 + 2 * j;
      const float* c0 = qb + (size_t)(32 + d) * N + mb;
      K0.u[j] = packbf2(c0[lo], c0[N + lo]);
      K1.u[j] = packbf2(c0[16 + lo], c0[N + 16 + lo]);
    }
    v8f S0 = {}, S1 = {};
    S0 = __builtin_amdgcn_wmma_f32_16x16x32_bf16(false, QA.v, false, K0.v, (short)0, S0, false, false);
    S1 = __builtin_amdgcn_wmma_f32_16x16x32_bf16(false, QA.v, false, K1.v, (short)0, S1, false, false);

    float knA = nk[(size_t)b * N + mb + lo];
    float knB = nk[(size_t)b * N + mb + 16 + lo];
#pragma unroll
    for (int r = 0; r < 8; ++r) {
      float s0 = S0[r] / (qnr[r] * knA + 1e-4f);
      float s1 = S1[r] / (qnr[r] * knB + 1e-4f);
      Sbuf[(hi * 8 + r) * 32 + lo]      = f2bf(s0);
      Sbuf[(hi * 8 + r) * 32 + 16 + lo] = f2bf(s1);
    }
    __syncthreads();
    v16bf sa = afrag16x32((const u32*)Sbuf, lane);   // 2x ds_load_b128
    __syncthreads();

    union { v16bf v; u32 u[8]; } V0, V1;
    const v4f* vp0 = (const v4f*)(qb + (size_t)(64 + lo) * N + mb + hi * 16);
    const v4f* vp1 = (const v4f*)(qb + (size_t)(80 + lo) * N + mb + hi * 16);
#pragma unroll
    for (int q = 0; q < 4; ++q) {
      v4f f0 = vp0[q], f1 = vp1[q];
      V0.u[2 * q]     = packbf2(f0[0], f0[1]);
      V0.u[2 * q + 1] = packbf2(f0[2], f0[3]);
      V1.u[2 * q]     = packbf2(f1[0], f1[1]);
      V1.u[2 * q + 1] = packbf2(f1[2], f1[3]);
    }
    O0 = __builtin_amdgcn_wmma_f32_16x16x32_bf16(false, sa, false, V0.v, (short)0, O0, false, false);
    O1 = __builtin_amdgcn_wmma_f32_16x16x32_bf16(false, sa, false, V1.v, (short)0, O1, false, false);
  }

  v4f w0, w1, w2, w3;
#pragma unroll
  for (int r = 0; r < 4; ++r) {
    w0[r] = O0[r]; w1[r] = O0[4 + r];
    w2[r] = O1[r]; w3[r] = O1[4 + r];
  }
  {
    v4f* p0 = (v4f*)(outp + ((size_t)b * 32 + lo) * N + n0 + hi * 8);
    p0[0] = w0; p0[1] = w1;
    v4f* p1 = (v4f*)(outp + ((size_t)b * 32 + 16 + lo) * N + n0 + hi * 8);
    p1[0] = w2; p1[1] = w3;
  }
}

// ---------------- global avg pool + FC ---------------------------------------
// FC weights are async-copied global->LDS (ASYNCcnt path) overlapped with the
// pooling reduction; s_wait_asynccnt + barrier before use.
__global__ __launch_bounds__(256)
void pool_fc(const float* __restrict__ x, const float* __restrict__ fcw,
             const float* __restrict__ fcb, float* __restrict__ out, int N) {
  int b = blockIdx.x, c = threadIdx.x;
  __shared__ float pooled[256];
  __shared__ __align__(16) float fcwLDS[4 * 256];

  {
    // per-lane 16B async copy: 256 threads x 4 floats = whole 4x256 matrix.
    // Generic LDS pointer low 32 bits == wave-relative LDS byte address.
    u32 ldsoff = (u32)(size_t)(void*)&fcwLDS[c * 4];
    unsigned long long gaddr = (unsigned long long)(size_t)(fcw + c * 4);
    asm volatile("global_load_async_to_lds_b128 %0, %1, off"
                 :: "v"(ldsoff), "v"(gaddr) : "memory");
  }

  const v4f* p4 = (const v4f*)(x + ((size_t)b * 256 + c) * N);
  float s = 0.f;
  for (int i = 0; i < N / 4; ++i) { v4f t = p4[i]; s += t[0] + t[1] + t[2] + t[3]; }
  pooled[c] = s / (float)N;

  asm volatile("s_wait_asynccnt 0" ::: "memory");
  __syncthreads();
  if (c < 4) {
    float acc = fcb[c];
    for (int j = 0; j < 256; ++j) acc += pooled[j] * fcwLDS[c * 256 + j];
    out[b * 4 + c] = acc;
  }
}

// =============================================================================
extern "C" void kernel_launch(void* const* d_in, const int* in_sizes, int n_in,
                              void* d_out, int out_size, void* d_ws, size_t ws_size,
                              hipStream_t stream) {
  (void)in_sizes; (void)n_in; (void)out_size; (void)ws_size;
  const float* img = (const float*)d_in[0];
  const float *ew[4], *eg[4], *eb[4], *em[4], *ev[4];
  for (int i = 0; i < 4; ++i) {            // enc_params: (w,(g,b,m,v)) x4
    ew[i] = (const float*)d_in[1 + 5 * i];
    eg[i] = (const float*)d_in[2 + 5 * i];
    eb[i] = (const float*)d_in[3 + 5 * i];
    em[i] = (const float*)d_in[4 + 5 * i];
    ev[i] = (const float*)d_in[5 + 5 * i];
  }
  const float* fcw = (const float*)d_in[149];
  const float* fcb = (const float*)d_in[150];

  // workspace arena (~200 MB peak)
  char* ws = (char*)d_ws;
  float* e0    = (float*)(ws);                     // 134 MB (conv1 out; reused as ffh)
  float* e1    = (float*)(ws + 134217728ull);      // 33.5 MB (pool buffers)
  float* xA    = (float*)(ws + 167772160ull);      // 16.8 MB (x)
  float* xB    = (float*)(ws + 184549376ull);      // 16.8 MB (x tmp)
  float* qkvb  = (float*)(ws + 201326592ull);      // 6.3 MB
  float* attnO = (float*)(ws + 207618048ull);      // 2.1 MB
  float* nq    = (float*)(ws + 209715200ull);      // 64 KB + 64 KB
  float* nk    = nq + 8 * 2048;
  float* ffh   = e0;                               // 67 MB alias

  const int Bn = 8, N = 2048;

  // ---- encoder ----
  { dim3 g(256 * 512 / 64, 2, Bn);   // Cin=3 -> generic small kernel
    conv3x3_small<<<g, 128, 0, stream>>>(img, ew[0], eg[0], eb[0], em[0], ev[0],
                                         e0, 3, 32, 256, 512, 9); }
  { int tot = Bn * 32 * 128 * 256;
    maxpool2<<<(tot + 255) / 256, 256, 0, stream>>>(e0, e1, 32, 256, 512, tot); }
  { dim3 g(128 * 256 / 64, 1, Bn);   // 32->64
    conv3x3_wmma<4><<<g, 128, 0, stream>>>(e1, ew[1], eg[1], eb[1], em[1], ev[1],
                                           nullptr, e0, 32, 64, 128, 256, 8); }
  { int tot = Bn * 64 * 64 * 128;
    maxpool2<<<(tot + 255) / 256, 256, 0, stream>>>(e0, e1, 64, 128, 256, tot); }
  { dim3 g(64 * 128 / 64, 2, Bn);    // 64->128
    conv3x3_wmma<4><<<g, 128, 0, stream>>>(e1, ew[2], eg[2], eb[2], em[2], ev[2],
                                           nullptr, e0, 64, 128, 64, 128, 7); }
  { int tot = Bn * 128 * 32 * 64;
    maxpool2<<<(tot + 255) / 256, 256, 0, stream>>>(e0, e1, 128, 64, 128, tot); }
  { dim3 g(32 * 64 / 64, 4, Bn);     // 128->256
    conv3x3_wmma<4><<<g, 128, 0, stream>>>(e1, ew[3], eg[3], eb[3], em[3], ev[3],
                                           nullptr, xA, 128, 256, 32, 64, 6); }

  // ---- transformer layers ----
  for (int l = 0; l < 8; ++l) {
    int base = 21 + 16 * l;
    const float* wqkv = (const float*)d_in[base + 0];
    const float* wout = (const float*)d_in[base + 1];
    const float* og = (const float*)d_in[base + 2];
    const float* ob = (const float*)d_in[base + 3];
    const float* om = (const float*)d_in[base + 4];
    const float* ov = (const float*)d_in[base + 5];
    const float* w1 = (const float*)d_in[base + 6];
    const float* g1 = (const float*)d_in[base + 7];
    const float* b1 = (const float*)d_in[base + 8];
    const float* m1 = (const float*)d_in[base + 9];
    const float* v1 = (const float*)d_in[base + 10];
    const float* w2 = (const float*)d_in[base + 11];
    const float* g2 = (const float*)d_in[base + 12];
    const float* b2 = (const float*)d_in[base + 13];
    const float* m2 = (const float*)d_in[base + 14];
    const float* v2 = (const float*)d_in[base + 15];

    { dim3 g(N / 64, 3, Bn);          // qkv conv3x3 (Cout=96 -> NT=2), no BN/ReLU
      conv3x3_wmma<2><<<g, 128, 0, stream>>>(xA, wqkv, nullptr, nullptr, nullptr, nullptr,
                                             nullptr, qkvb, 256, 96, 32, 64, 6); }
    { int tot = 2 * Bn * N;
      qk_norms<<<(tot + 255) / 256, 256, 0, stream>>>(qkvb, nq, nk, Bn, N); }
    { dim3 g(N / 16, Bn);
      attention_wmma<<<g, 32, 0, stream>>>(qkvb, nq, nk, attnO, N); }
    { dim3 g(N / 64, 4, Bn);          // to_out proj + BN/ReLU + residual(x)
      conv1x1_wmma<4><<<g, 128, 0, stream>>>(attnO, wout, og, ob, om, ov, xA, xB, 32, 256, N); }
    { dim3 g(N / 64, 16, Bn);         // ff conv1 + BN/ReLU
      conv1x1_wmma<4><<<g, 128, 0, stream>>>(xB, w1, g1, b1, m1, v1, nullptr, ffh, 256, 1024, N); }
    { dim3 g(N / 64, 4, Bn);          // ff conv2 + BN/ReLU + residual(xB) -> xA
      conv1x1_wmma<4><<<g, 128, 0, stream>>>(ffh, w2, g2, b2, m2, v2, xB, xA, 1024, 256, N); }
  }

  pool_fc<<<Bn, 256, 0, stream>>>(xA, fcw, fcb, (float*)d_out, N);
}
